// GIN_6536940225142
// MI455X (gfx1250) — matrix operand order repaired and checked by reading, not compiled, not run on hardware
//
#include <hip/hip_runtime.h>

// ---- problem constants (from reference) ----
#define Nn    100000
#define Ee    300000
#define Gg    2000
#define DIN   128
#define EIN   16
#define Hh    300
#define Rr    1024
#define DEPTHN 3
#define HS    320   // padded stride for h/z/t/e/mean (K padded to mult of 32)
#define AS    304   // stride for agg / gsum (mult of 16)
#define EKP   32    // edge-attr K padded to 32

typedef __attribute__((ext_vector_type(16))) __bf16 v16bf;
typedef __attribute__((ext_vector_type(8)))  __bf16 v8bf;
typedef __attribute__((ext_vector_type(8)))  float  v8f;

// ---------------------------------------------------------------------------
// Convert f32 [rows, scols] -> bf16 [rows, dstride], zero-filling pad columns.
// ---------------------------------------------------------------------------
__global__ void cvt_pad_bf16(const float* __restrict__ src, __bf16* __restrict__ dst,
                             int rows, int scols, int dstride) {
  size_t i = (size_t)blockIdx.x * blockDim.x + threadIdx.x;
  size_t total = (size_t)rows * dstride;
  if (i >= total) return;
  int c = (int)(i % dstride);
  size_t r = i / dstride;
  dst[i] = (c < scols) ? (__bf16)src[r * (size_t)scols + c] : (__bf16)0.0f;
}

// ---------------------------------------------------------------------------
// Pack a f32 weight matrix W[K, Ncols] (row-major) into WMMA B fragments:
//   layout [nt][kt][lane][elt] ; lane holds column nt*16 + lane%16,
//   element e -> k = kt*32 + (lane/16)*16 + e   (contiguous k order, same as the A-fragment loads)
// Zero-fills out-of-range K / columns (so padded n-tiles are all-zero).
// ---------------------------------------------------------------------------
__global__ void pack_b_bf16(const float* __restrict__ W, __bf16* __restrict__ Bp,
                            int K, int Ncols, int KT) {
  int nt = blockIdx.x, kt = blockIdx.y, lane = threadIdx.x;
  int n = nt * 16 + (lane & 15);
  __bf16* out = Bp + ((size_t)(nt * KT + kt) * 32 + lane) * 16;
#pragma unroll
  for (int e = 0; e < 16; ++e) {
    int k = kt * 32 + ((lane >> 4) << 4) + e;
    float v = (k < K && n < Ncols) ? W[(size_t)k * Ncols + n] : 0.0f;
    out[e] = (__bf16)v;
  }
}

// ---------------------------------------------------------------------------
// WMMA GEMM: C = act(A[M,Kpad]_bf16 @ B_packed + bias)
//   One wave owns a 16x64 output strip (1 m-tile x 4 n-tiles); the A fragment
//   is loaded once per k-step and reused by 4 back-to-back WMMAs. NT is
//   required to be a multiple of 4 (packed B zero-padded), so the inner loop
//   is completely branch-free and KT is a compile-time constant (full unroll,
//   immediate-offset global_load_b128). Block = 128 threads = 4 waves.
//   act: 0=identity, 1=relu, 2=prelu(*prelu_a)
//   Cbf!=null -> bf16 out, stride ldc, zero-filled pad cols (col>=ncols)
//   else      -> f32 out
// ---------------------------------------------------------------------------
template <int KT>
__global__ void wmma_gemm(const __bf16* __restrict__ A, const __bf16* __restrict__ Bp,
                          const float* __restrict__ bias,
                          __bf16* __restrict__ Cbf, float* __restrict__ Cf,
                          int mtiles, int lda, int ldc, int ncols,
                          int act, const float* __restrict__ prelu_a) {
  int wave = threadIdx.x >> 5;
  int lane = threadIdx.x & 31;
  int mtile = blockIdx.x * (blockDim.x >> 5) + wave;
  int nt0   = blockIdx.y * 4;
  if (mtile >= mtiles) return;               // wave-uniform: EXEC stays all-ones

  // A fragment: row m0+lane%16, one contiguous 16-elt run (two 16B loads) per k-step
  int m0  = mtile << 4;
  int row = m0 + (lane & 15);
  int ksel = (lane >> 4) << 4;               // 0 or 16
  const __bf16* Arow  = A + (size_t)row * lda + ksel;
  const __bf16* Bbase = Bp + (size_t)nt0 * KT * 512 + (size_t)lane * 16;

  v8f acc0 = {}, acc1 = {}, acc2 = {}, acc3 = {};
#pragma unroll
  for (int kt = 0; kt < KT; ++kt) {
    v8bf alo = *(const v8bf*)(Arow + kt * 32);        // K = k0+ksel .. +7
    v8bf ahi = *(const v8bf*)(Arow + kt * 32 + 8);    // K = k0+ksel+8 .. +15
    v16bf a;
#pragma unroll
    for (int i = 0; i < 8; ++i) { a[i] = alo[i]; a[8 + i] = ahi[i]; }

#pragma unroll
    for (int j = 0; j < 4; ++j) {
      const __bf16* bf = Bbase + (size_t)(j * KT + kt) * 512;
      v8bf blo = *(const v8bf*)(bf);
      v8bf bhi = *(const v8bf*)(bf + 8);
      v16bf b;
#pragma unroll
      for (int i = 0; i < 8; ++i) { b[i] = blo[i]; b[8 + i] = bhi[i]; }
      v8f* accp = (j == 0) ? &acc0 : (j == 1) ? &acc1 : (j == 2) ? &acc2 : &acc3;
      *accp = __builtin_amdgcn_wmma_f32_16x16x32_bf16(false, a, false, b,
                                                      (short)0, *accp, false, false);
    }
  }

  float pa = (act == 2) ? *prelu_a : 0.0f;
  v8f accs[4] = {acc0, acc1, acc2, acc3};
#pragma unroll
  for (int j = 0; j < 4; ++j) {
    int col = ((nt0 + j) << 4) + (lane & 15);        // D-tile: N = lane%16
    float bv = (bias && col < ncols) ? bias[col] : 0.0f;
#pragma unroll
    for (int v = 0; v < 8; ++v) {
      float x = accs[j][v] + bv;
      if (act == 1)      x = x > 0.0f ? x : 0.0f;
      else if (act == 2) x = x >= 0.0f ? x : pa * x;
      int r = m0 + v + ((lane >> 4) << 3);           // D-tile: M = v + 8*(lane/16)
      if (Cbf) {
        Cbf[(size_t)r * ldc + col] = (col < ncols) ? (__bf16)x : (__bf16)0.0f;
      } else if (col < ncols) {
        Cf[(size_t)r * ldc + col] = x;
      }
    }
  }
}

// ---------------------------------------------------------------------------
// Fused message pass: msg = relu(h[src] + e), atomic scatter-add into agg[dst].
// One wave per edge; lane strides over the 300 features.
// ---------------------------------------------------------------------------
__global__ void edge_scatter(const __bf16* __restrict__ h, const __bf16* __restrict__ e,
                             const int* __restrict__ src, const int* __restrict__ dst,
                             float* __restrict__ agg) {
  int gw   = (int)(((size_t)blockIdx.x * blockDim.x + threadIdx.x) >> 5);
  int lane = threadIdx.x & 31;
  if (gw >= Ee) return;
  int s = src[gw], d = dst[gw];
  const __bf16* hr = h + (size_t)s * HS;
  const __bf16* er = e + (size_t)gw * HS;
  float* ar = agg + (size_t)d * AS;
  for (int c = lane; c < Hh; c += 32) {
    float m = (float)hr[c] + (float)er[c];
    m = m > 0.0f ? m : 0.0f;
    atomicAdd(&ar[c], m);
  }
}

// ---------------------------------------------------------------------------
// z = h + agg   (bf16 out, zero pad cols) — GINEConv eps=0 combine
// ---------------------------------------------------------------------------
__global__ void combine_hagg(const __bf16* __restrict__ h, const float* __restrict__ agg,
                             __bf16* __restrict__ z) {
  size_t i = (size_t)blockIdx.x * blockDim.x + threadIdx.x;
  if (i >= (size_t)Nn * HS) return;
  int c = (int)(i % HS);
  size_t n = i / HS;
  float v = (c < Hh) ? ((float)h[i] + agg[n * AS + c]) : 0.0f;
  z[i] = (__bf16)v;
}

// ---------------------------------------------------------------------------
// Per-graph segment sums (wave per node) + count.
// ---------------------------------------------------------------------------
__global__ void seg_sum(const __bf16* __restrict__ h, const int* __restrict__ gid,
                        float* __restrict__ gsum, float* __restrict__ gcnt) {
  int gw   = (int)(((size_t)blockIdx.x * blockDim.x + threadIdx.x) >> 5);
  int lane = threadIdx.x & 31;
  if (gw >= Nn) return;
  int g = gid[gw];
  const __bf16* hr = h + (size_t)gw * HS;
  float* sr = gsum + (size_t)g * AS;
  for (int c = lane; c < Hh; c += 32) atomicAdd(&sr[c], (float)hr[c]);
  if (lane == 0) atomicAdd(&gcnt[g], 1.0f);
}

__global__ void seg_mean(const float* __restrict__ gsum, const float* __restrict__ gcnt,
                         __bf16* __restrict__ mean) {
  size_t i = (size_t)blockIdx.x * blockDim.x + threadIdx.x;
  if (i >= (size_t)Gg * HS) return;
  int c = (int)(i % HS);
  size_t g = i / HS;
  float v = (c < Hh) ? (gsum[g * AS + c] / fmaxf(gcnt[g], 1.0f)) : 0.0f;
  mean[i] = (__bf16)v;
}

// ---------------------------------------------------------------------------
// host launch
// ---------------------------------------------------------------------------
template <int KT>
static inline void gemm_launch(const __bf16* A, const __bf16* Bp, const float* bias,
                               __bf16* Cbf, float* Cf, int mtiles, int lda,
                               int ldc, int ncols, int NT, int act, const float* pa,
                               hipStream_t s) {
  dim3 grid((mtiles + 3) / 4, NT / 4);       // NT is always a multiple of 4
  hipLaunchKernelGGL((wmma_gemm<KT>), grid, dim3(128), 0, s,
                     A, Bp, bias, Cbf, Cf, mtiles, lda, ldc, ncols, act, pa);
}

extern "C" void kernel_launch(void* const* d_in, const int* in_sizes, int n_in,
                              void* d_out, int out_size, void* d_ws, size_t ws_size,
                              hipStream_t stream) {
  const float* node_attr = (const float*)d_in[0];
  const float* edge_attr = (const float*)d_in[1];
  const int*   src       = (const int*)d_in[2];
  const int*   dst       = (const int*)d_in[3];
  const int*   gids      = (const int*)d_in[4];
  const float* Wn  = (const float*)d_in[5];
  const float* bn  = (const float*)d_in[6];
  const float* We  = (const float*)d_in[7];
  const float* be  = (const float*)d_in[8];
  const float* Wg1 = (const float*)d_in[9];
  const float* bg1 = (const float*)d_in[10];
  const float* Wg2 = (const float*)d_in[11];
  const float* bg2 = (const float*)d_in[12];
  const float* Wsp = (const float*)d_in[13];
  const float* bsp = (const float*)d_in[14];
  const float* pa  = (const float*)d_in[15];
  float* out = (float*)d_out;

  // ---- workspace carve-up ----
  char* ws = (char*)d_ws;
  size_t off = 0;
  auto alloc = [&](size_t bytes) -> void* {
    void* p = ws + off;
    off = (off + bytes + 255) & ~(size_t)255;
    return p;
  };
  __bf16* x_bf  = (__bf16*)alloc((size_t)Nn * DIN * 2);
  __bf16* ea_bf = (__bf16*)alloc((size_t)Ee * EKP * 2);
  __bf16* e_bf  = (__bf16*)alloc((size_t)Ee * HS * 2);
  __bf16* h_bf  = (__bf16*)alloc((size_t)Nn * HS * 2);
  __bf16* z_bf  = (__bf16*)alloc((size_t)Nn * HS * 2);
  __bf16* t_bf  = (__bf16*)alloc((size_t)Nn * HS * 2);
  float*  agg   = (float*)alloc((size_t)Nn * AS * 4);
  float*  gsum  = (float*)alloc((size_t)Gg * AS * 4);
  float*  gcnt  = (float*)alloc((size_t)Gg * 4);
  __bf16* mn_bf = (__bf16*)alloc((size_t)Gg * HS * 2);
  // packed weight fragments: bytes = NT*KT*1024; all NT multiples of 4
  const int NT_H = HS / 16, KT_H = HS / 32;        // 20, 10
  __bf16* Wn_p  = (__bf16*)alloc((size_t)NT_H * (DIN / 32) * 1024);
  __bf16* We_p  = (__bf16*)alloc((size_t)NT_H * 1 * 1024);
  __bf16* Wg1_p = (__bf16*)alloc((size_t)3 * NT_H * KT_H * 1024);
  __bf16* Wg2_p = (__bf16*)alloc((size_t)3 * NT_H * KT_H * 1024);
  __bf16* Wsp_p = (__bf16*)alloc((size_t)(Rr / 16) * KT_H * 1024);

  // ---- stage inputs / weights into bf16 + WMMA fragment layout ----
  {
    size_t t = (size_t)Nn * DIN;
    hipLaunchKernelGGL(cvt_pad_bf16, dim3((t + 255) / 256), dim3(256), 0, stream,
                       node_attr, x_bf, Nn, DIN, DIN);
    t = (size_t)Ee * EKP;
    hipLaunchKernelGGL(cvt_pad_bf16, dim3((t + 255) / 256), dim3(256), 0, stream,
                       edge_attr, ea_bf, Ee, EIN, EKP);
  }
  hipLaunchKernelGGL(pack_b_bf16, dim3(NT_H, DIN / 32), dim3(32), 0, stream,
                     Wn, Wn_p, DIN, Hh, DIN / 32);
  hipLaunchKernelGGL(pack_b_bf16, dim3(NT_H, 1), dim3(32), 0, stream,
                     We, We_p, EIN, Hh, 1);
  for (int i = 0; i < DEPTHN; ++i) {
    hipLaunchKernelGGL(pack_b_bf16, dim3(NT_H, KT_H), dim3(32), 0, stream,
                       Wg1 + (size_t)i * Hh * Hh, Wg1_p + (size_t)i * NT_H * KT_H * 512,
                       Hh, Hh, KT_H);
    hipLaunchKernelGGL(pack_b_bf16, dim3(NT_H, KT_H), dim3(32), 0, stream,
                       Wg2 + (size_t)i * Hh * Hh, Wg2_p + (size_t)i * NT_H * KT_H * 512,
                       Hh, Hh, KT_H);
  }
  hipLaunchKernelGGL(pack_b_bf16, dim3(Rr / 16, KT_H), dim3(32), 0, stream,
                     Wsp, Wsp_p, Hh, Rr, KT_H);

  // ---- projections ----
  // h = relu(node_attr @ Wn + bn)
  gemm_launch<DIN / 32>(x_bf, Wn_p, bn, h_bf, nullptr, Nn / 16, DIN, HS, Hh,
                        NT_H, 1, nullptr, stream);
  // e = edge_attr @ We + be
  gemm_launch<1>(ea_bf, We_p, be, e_bf, nullptr, Ee / 16, EKP, HS, Hh,
                 NT_H, 0, nullptr, stream);

  // ---- GINEConv layers ----
  for (int i = 0; i < DEPTHN; ++i) {
    hipMemsetAsync(agg, 0, (size_t)Nn * AS * 4, stream);
    hipLaunchKernelGGL(edge_scatter, dim3((Ee + 7) / 8), dim3(256), 0, stream,
                       h_bf, e_bf, src, dst, agg);
    {
      size_t t = (size_t)Nn * HS;
      hipLaunchKernelGGL(combine_hagg, dim3((t + 255) / 256), dim3(256), 0, stream,
                         h_bf, agg, z_bf);
    }
    // t = relu(z @ Wg1 + bg1)
    gemm_launch<HS / 32>(z_bf, Wg1_p + (size_t)i * NT_H * KT_H * 512,
                         bg1 + (size_t)i * Hh, t_bf, nullptr, Nn / 16, HS, HS, Hh,
                         NT_H, 1, nullptr, stream);
    // h = act(t @ Wg2 + bg2), relu except last layer
    gemm_launch<HS / 32>(t_bf, Wg2_p + (size_t)i * NT_H * KT_H * 512,
                         bg2 + (size_t)i * Hh, h_bf, nullptr, Nn / 16, HS, HS, Hh,
                         NT_H, (i < DEPTHN - 1) ? 1 : 0, nullptr, stream);
  }

  // ---- readout: per-graph mean ----
  hipMemsetAsync(gsum, 0, (size_t)Gg * AS * 4, stream);
  hipMemsetAsync(gcnt, 0, (size_t)Gg * 4, stream);
  hipLaunchKernelGGL(seg_sum, dim3((Nn + 7) / 8), dim3(256), 0, stream,
                     h_bf, gids, gsum, gcnt);
  {
    size_t t = (size_t)Gg * HS;
    hipLaunchKernelGGL(seg_mean, dim3((t + 255) / 256), dim3(256), 0, stream,
                       gsum, gcnt, mn_bf);
  }

  // ---- sparsify: out = prelu(mean @ Wsp + bsp) ----
  gemm_launch<HS / 32>(mn_bf, Wsp_p, bsp, nullptr, out, Gg / 16, HS, Rr, Rr,
                       Rr / 16, 2, pa, stream);
}
